// GRU_7121055776972
// MI455X (gfx1250) — compile-verified, matrix operand order checked by reading, not run on hardware
//
#include <hip/hip_runtime.h>
#include <hip/hip_bf16.h>

#define SEQ    512
#define BS     128
#define IN_DIM 512
#define H_DIM  1024
#define D_DIM  1536   /* H_DIM + IN_DIM */
#define NG     2048   /* 2 * H_DIM (z|r fused) */
#define NT     4      /* N-tiles (16 cols each) per wave: 16x64 wave tile */

typedef __attribute__((ext_vector_type(16))) __bf16 v16bf;
typedef __attribute__((ext_vector_type(8)))  __bf16 v8bf;
typedef __attribute__((ext_vector_type(8)))  float  v8f;

__device__ __forceinline__ unsigned short f2bf_bits(float f) {
    // round-to-nearest-even fp32 -> bf16
    unsigned u = __float_as_uint(f);
    return (unsigned short)((u + 0x7FFFu + ((u >> 16) & 1u)) >> 16);
}

// Build a 16x32 bf16 WMMA A/B fragment per the CDNA5 ISA layout:
// lanes 0-15 hold row l16 with K = {kk..kk+7, kk+16..kk+23},
// lanes 16-31 hold row l16 with K = {kk+8..kk+15, kk+24..kk+31}.
__device__ __forceinline__ v16bf load_frag(const unsigned short* rowbase, int half) {
    const v8bf lo = *(const v8bf*)(rowbase + half * 8);
    const v8bf hi = *(const v8bf*)(rowbase + half * 8 + 16);
    return __builtin_shufflevector(lo, hi, 0,1,2,3,4,5,6,7,8,9,10,11,12,13,14,15);
}

// ---------------- one-time prep kernels ----------------

__global__ __launch_bounds__(256) void k_cvt_bf16(const float* __restrict__ in,
                                                  unsigned short* __restrict__ out, int n) {
    for (int i = blockIdx.x * blockDim.x + threadIdx.x; i < n; i += gridDim.x * blockDim.x)
        out[i] = f2bf_bits(in[i]);
}

// W: [D_DIM, H_DIM] (row = input feature k) -> out: [H_DIM, D_DIM] bf16 (N-major)
__global__ __launch_bounds__(256) void k_pack_w(const float* __restrict__ W,
                                                unsigned short* __restrict__ out) {
    const int total = H_DIM * D_DIM;
    for (int i = blockIdx.x * blockDim.x + threadIdx.x; i < total; i += gridDim.x * blockDim.x) {
        const int nrow = i / D_DIM;
        const int k    = i - nrow * D_DIM;
        out[i] = f2bf_bits(W[(size_t)k * H_DIM + nrow]);
    }
}

__global__ __launch_bounds__(256) void k_init_h(const float* __restrict__ h0,
                                                float* __restrict__ hF,
                                                unsigned short* __restrict__ hB, int n) {
    for (int i = blockIdx.x * blockDim.x + threadIdx.x; i < n; i += gridDim.x * blockDim.x) {
        const float v = h0[i];
        hF[i] = v;
        hB[i] = f2bf_bits(v);
    }
}

// ---------------- per-step kernels ----------------

// logits[128, 2048] = [h | x] @ [Wz|Wr]^T(N-major) + [bz|br]
// Each wave computes a 16(M) x 64(N) tile: one A fragment feeds 4 WMMAs.
__global__ __launch_bounds__(256) void gru_gates(
    const unsigned short* __restrict__ hB,     // [BS, H_DIM] bf16
    const unsigned short* __restrict__ xB,     // [BS, IN_DIM] bf16 (step t)
    const unsigned short* __restrict__ WzrB,   // [NG, D_DIM] bf16
    const float* __restrict__ bz, const float* __restrict__ br,
    float* __restrict__ logits)                // [BS, NG]
{
    const int lane  = threadIdx.x & 31;
    const int wave  = threadIdx.x >> 5;        // 8 waves = 8 M-tiles (full M=128)
    const int half  = lane >> 4;
    const int l16   = lane & 15;
    const int nbase = blockIdx.x * (NT * 16);  // 64-wide N group
    const int mtile = wave;

    const unsigned short* wr[NT];
    #pragma unroll
    for (int j = 0; j < NT; ++j)
        wr[j] = WzrB + (size_t)(nbase + j * 16 + l16) * D_DIM;

    const unsigned short* hrow = hB + (size_t)(mtile * 16 + l16) * H_DIM;
    const unsigned short* xrow = xB + (size_t)(mtile * 16 + l16) * IN_DIM;

    v8f acc[NT] = {{}, {}, {}, {}};
    #pragma unroll 2
    for (int kk = 0; kk < D_DIM; kk += 32) {
        const unsigned short* arow = (kk < H_DIM) ? (hrow + kk) : (xrow + (kk - H_DIM));
        const v16bf a = load_frag(arow, half);
        #pragma unroll
        for (int j = 0; j < NT; ++j) {
            const v16bf b = load_frag(wr[j] + kk, half);
            acc[j] = __builtin_amdgcn_wmma_f32_16x16x32_bf16(false, a, false, b,
                                                             (short)0, acc[j], false, false);
        }
    }
    #pragma unroll
    for (int j = 0; j < NT; ++j) {
        const int n = nbase + j * 16 + l16;
        const float bias = (n < H_DIM) ? bz[n] : br[n - H_DIM];
        #pragma unroll
        for (int v = 0; v < 8; ++v) {
            const int mm = mtile * 16 + v + 8 * half;
            logits[(size_t)mm * NG + n] = acc[j][v] + bias;
        }
    }
}

// per-row softmax of z and r logits; emits z (f32) and bf16(r*h)
__global__ __launch_bounds__(256) void gru_softmax(
    const float* __restrict__ logits,          // [BS, NG]
    const float* __restrict__ hF,              // [BS, H_DIM]
    float* __restrict__ zF,                    // [BS, H_DIM]
    unsigned short* __restrict__ rhB)          // [BS, H_DIM] bf16
{
    const int m    = blockIdx.x;
    const int tid  = threadIdx.x;
    const int lane = tid & 31;
    const int wv   = tid >> 5;
    __shared__ float sred[8];

    // ---------------- z gate ----------------
    const float* Lz = logits + (size_t)m * NG;
    float a0 = Lz[tid], a1 = Lz[tid + 256], a2 = Lz[tid + 512], a3 = Lz[tid + 768];
    float mx = fmaxf(fmaxf(a0, a1), fmaxf(a2, a3));
    #pragma unroll
    for (int off = 16; off; off >>= 1) mx = fmaxf(mx, __shfl_xor(mx, off, 32));
    if (lane == 0) sred[wv] = mx;
    __syncthreads();
    float bmax = sred[0];
    #pragma unroll
    for (int i = 1; i < 8; ++i) bmax = fmaxf(bmax, sred[i]);
    __syncthreads();
    a0 = __expf(a0 - bmax); a1 = __expf(a1 - bmax);
    a2 = __expf(a2 - bmax); a3 = __expf(a3 - bmax);
    float sum = a0 + a1 + a2 + a3;
    #pragma unroll
    for (int off = 16; off; off >>= 1) sum += __shfl_xor(sum, off, 32);
    if (lane == 0) sred[wv] = sum;
    __syncthreads();
    float tot = 0.f;
    #pragma unroll
    for (int i = 0; i < 8; ++i) tot += sred[i];
    __syncthreads();
    const float inv = 1.0f / tot;
    float* Zo = zF + (size_t)m * H_DIM;
    Zo[tid] = a0 * inv; Zo[tid + 256] = a1 * inv;
    Zo[tid + 512] = a2 * inv; Zo[tid + 768] = a3 * inv;

    // ---------------- r gate, fused r*h ----------------
    const float* Lr = Lz + H_DIM;
    float r0 = Lr[tid], r1 = Lr[tid + 256], r2 = Lr[tid + 512], r3 = Lr[tid + 768];
    float mr = fmaxf(fmaxf(r0, r1), fmaxf(r2, r3));
    #pragma unroll
    for (int off = 16; off; off >>= 1) mr = fmaxf(mr, __shfl_xor(mr, off, 32));
    if (lane == 0) sred[wv] = mr;
    __syncthreads();
    float bmr = sred[0];
    #pragma unroll
    for (int i = 1; i < 8; ++i) bmr = fmaxf(bmr, sred[i]);
    __syncthreads();
    r0 = __expf(r0 - bmr); r1 = __expf(r1 - bmr);
    r2 = __expf(r2 - bmr); r3 = __expf(r3 - bmr);
    float sr = r0 + r1 + r2 + r3;
    #pragma unroll
    for (int off = 16; off; off >>= 1) sr += __shfl_xor(sr, off, 32);
    if (lane == 0) sred[wv] = sr;
    __syncthreads();
    float totr = 0.f;
    #pragma unroll
    for (int i = 0; i < 8; ++i) totr += sred[i];
    const float invr = 1.0f / totr;
    const float* Hr = hF + (size_t)m * H_DIM;
    unsigned short* Ro = rhB + (size_t)m * H_DIM;
    Ro[tid]       = f2bf_bits(r0 * invr * Hr[tid]);
    Ro[tid + 256] = f2bf_bits(r1 * invr * Hr[tid + 256]);
    Ro[tid + 512] = f2bf_bits(r2 * invr * Hr[tid + 512]);
    Ro[tid + 768] = f2bf_bits(r3 * invr * Hr[tid + 768]);
}

// candidate GEMM + state update: h' = h + z*(tanh([r*h|x]@Wh + bh) - h)
__global__ __launch_bounds__(256) void gru_cand(
    const unsigned short* __restrict__ rhB,    // [BS, H_DIM] bf16
    const unsigned short* __restrict__ xB,     // [BS, IN_DIM] bf16 (step t)
    const unsigned short* __restrict__ WhB,    // [H_DIM, D_DIM] bf16
    const float* __restrict__ bh,
    const float* __restrict__ zF,              // [BS, H_DIM]
    float* __restrict__ hF,                    // in/out [BS, H_DIM]
    unsigned short* __restrict__ hB,           // out [BS, H_DIM] bf16
    float* __restrict__ outT)                  // out[t]: [BS, H_DIM]
{
    const int lane  = threadIdx.x & 31;
    const int wave  = threadIdx.x >> 5;
    const int half  = lane >> 4;
    const int l16   = lane & 15;
    const int nbase = blockIdx.x * (NT * 16);
    const int mtile = wave;

    const unsigned short* wr[NT];
    #pragma unroll
    for (int j = 0; j < NT; ++j)
        wr[j] = WhB + (size_t)(nbase + j * 16 + l16) * D_DIM;

    const unsigned short* rrow = rhB + (size_t)(mtile * 16 + l16) * H_DIM;
    const unsigned short* xrow = xB  + (size_t)(mtile * 16 + l16) * IN_DIM;

    v8f acc[NT] = {{}, {}, {}, {}};
    #pragma unroll 2
    for (int kk = 0; kk < D_DIM; kk += 32) {
        const unsigned short* arow = (kk < H_DIM) ? (rrow + kk) : (xrow + (kk - H_DIM));
        const v16bf a = load_frag(arow, half);
        #pragma unroll
        for (int j = 0; j < NT; ++j) {
            const v16bf b = load_frag(wr[j] + kk, half);
            acc[j] = __builtin_amdgcn_wmma_f32_16x16x32_bf16(false, a, false, b,
                                                             (short)0, acc[j], false, false);
        }
    }
    #pragma unroll
    for (int j = 0; j < NT; ++j) {
        const int n = nbase + j * 16 + l16;
        const float bias = bh[n];
        #pragma unroll
        for (int v = 0; v < 8; ++v) {
            const int mm = mtile * 16 + v + 8 * half;
            const size_t idx = (size_t)mm * H_DIM + n;
            const float c  = tanhf(acc[j][v] + bias);
            const float z  = zF[idx];
            const float h  = hF[idx];
            const float hn = h + z * (c - h);
            outT[idx] = hn;
            hF[idx]   = hn;
            hB[idx]   = f2bf_bits(hn);
        }
    }
}

// ---------------- host launcher ----------------

extern "C" void kernel_launch(void* const* d_in, const int* in_sizes, int n_in,
                              void* d_out, int out_size, void* d_ws, size_t ws_size,
                              hipStream_t stream) {
    const float* x  = (const float*)d_in[0];
    const float* h0 = (const float*)d_in[1];
    const float* Wz = (const float*)d_in[2];
    const float* bz = (const float*)d_in[3];
    const float* Wr = (const float*)d_in[4];
    const float* br = (const float*)d_in[5];
    const float* Wh = (const float*)d_in[6];
    const float* bh = (const float*)d_in[7];
    float* out = (float*)d_out;

    char* wsb = (char*)d_ws;
    size_t off = 0;
    auto take = [&](size_t bytes) -> char* {
        char* r = wsb + off;
        off += (bytes + 255) & ~(size_t)255;
        return r;
    };
    unsigned short* xB     = (unsigned short*)take((size_t)SEQ * BS * IN_DIM * 2);
    unsigned short* WzrB   = (unsigned short*)take((size_t)NG * D_DIM * 2);
    unsigned short* WhB    = (unsigned short*)take((size_t)H_DIM * D_DIM * 2);
    float*          hF     = (float*)take((size_t)BS * H_DIM * 4);
    float*          zF     = (float*)take((size_t)BS * H_DIM * 4);
    unsigned short* rhB    = (unsigned short*)take((size_t)BS * H_DIM * 2);
    unsigned short* hB     = (unsigned short*)take((size_t)BS * H_DIM * 2);
    float*          logits = (float*)take((size_t)BS * NG * 4);

    // one-time prep: bf16 copies + N-major weight packs + state init
    k_cvt_bf16<<<2048, 256, 0, stream>>>(x, xB, SEQ * BS * IN_DIM);
    k_pack_w  <<<1536, 256, 0, stream>>>(Wz, WzrB);
    k_pack_w  <<<1536, 256, 0, stream>>>(Wr, WzrB + (size_t)H_DIM * D_DIM);
    k_pack_w  <<<1536, 256, 0, stream>>>(Wh, WhB);
    k_init_h  <<<512, 256, 0, stream>>>(h0, hF, hB, BS * H_DIM);

    // sequential scan: 3 kernels per step, ordered by the stream
    for (int t = 0; t < SEQ; ++t) {
        const unsigned short* xt = xB + (size_t)t * BS * IN_DIM;
        gru_gates  <<<NG / (NT * 16),    256, 0, stream>>>(hB, xt, WzrB, bz, br, logits);
        gru_softmax<<<BS,                256, 0, stream>>>(logits, hF, zF, rhB);
        gru_cand   <<<H_DIM / (NT * 16), 256, 0, stream>>>(rhB, xt, WhB, bh, zF, hF, hB,
                                                           out + (size_t)t * BS * H_DIM);
    }
}
